// MyGATClassifier_35485019799703
// MI455X (gfx1250) — matrix operand order repaired and checked by reading, not compiled, not run on hardware
//
#include <hip/hip_runtime.h>
#include <hip/hip_bf16.h>
#include <math.h>
#include <stdint.h>

// ---------------------------------------------------------------------------
// Types for CDNA5 WMMA (wave32): v_wmma_f32_16x16x32_bf16
// ---------------------------------------------------------------------------
typedef __bf16 bf16_t;
typedef __attribute__((ext_vector_type(16))) __bf16 bf16x16;
typedef __attribute__((ext_vector_type(8)))  __bf16 bf16x8;
typedef __attribute__((ext_vector_type(8)))  float  f32x8;

#define HID 128   // hidden = H*D = heads(4) * dim(32)
#define NHEAD 4
#define HDIM 32
#define NCLS 32   // V (node types) == classifier per-head dim

// ---------------------------------------------------------------------------
// CDNA5 async memory->LDS copy (ASYNCcnt-tracked), 16 bytes per lane.
// lds_addr: wave-relative LDS byte address (low 32 bits of a generic pointer
// to __shared__); gaddr: 64-bit global VA.
// ---------------------------------------------------------------------------
__device__ __forceinline__ void async_copy16(unsigned lds_addr, const void* gaddr) {
    asm volatile("global_load_async_to_lds_b128 %0, %1, off"
                 :: "v"(lds_addr), "v"((unsigned long long)(uintptr_t)gaddr)
                 : "memory");
}
__device__ __forceinline__ void wait_async0() {
    asm volatile("s_wait_asynccnt 0" ::: "memory");
}

// ---------------------------------------------------------------------------
// Utility: float atomic max via CAS (portable; gfx1250 also has native f32 max)
// ---------------------------------------------------------------------------
__device__ __forceinline__ void atomicMaxF(float* addr, float val) {
    int old = __float_as_int(*addr);
    while (val > __int_as_float(old)) {
        int assumed = old;
        old = atomicCAS((int*)addr, assumed, __float_as_int(val));
        if (old == assumed) break;
    }
}

// ---------------------------------------------------------------------------
// Fill a float buffer with a constant (init m = -inf, s = 0, accum = 0)
// ---------------------------------------------------------------------------
__global__ void k_fill_f32(float* __restrict__ p, float v, int n) {
    int i = blockIdx.x * blockDim.x + threadIdx.x;
    if (i < n) p[i] = v;
}

// ---------------------------------------------------------------------------
// Convert a 128x128 f32 weight (K x Ncol, row-major) into bf16 B^T (col-major
// of original => Bt[col][k]) so WMMA B-fragments are contiguous in LDS.
// ---------------------------------------------------------------------------
__global__ void k_conv_weight(const float* __restrict__ W, bf16_t* __restrict__ Bt) {
    int tid = blockIdx.x * blockDim.x + threadIdx.x;   // 128*128 threads
    int k = tid >> 7, c = tid & 127;
    Bt[(size_t)c * HID + k] = (bf16_t)W[(size_t)k * HID + c];
}

// ---------------------------------------------------------------------------
// Gather embedding rows -> bf16 activations (only GEMMs consume h directly)
// ---------------------------------------------------------------------------
__global__ void k_gather_emb(const int* __restrict__ feat, const float* __restrict__ emb,
                             bf16_t* __restrict__ h, int N) {
    int tid = blockIdx.x * blockDim.x + threadIdx.x;
    int n = tid >> 7, c = tid & 127;
    if (n < N) h[(size_t)n * HID + c] = (bf16_t)emb[(size_t)feat[n] * HID + c];
}

// ---------------------------------------------------------------------------
// Shared GEMM machinery. Block: 128 threads = 4 waves. M-tile = 64 rows;
// each wave: 16 rows x 8 col-tiles x 4 k-steps = 32 WMMAs.
// ---------------------------------------------------------------------------
#define GEMM_MT  64
#define GEMM_LDA 136   // padded LDS row (bf16 elems); 136*2=272 B, 16B aligned

union Frag { bf16x16 v; bf16x8 h[2]; };

// stage A tile (64 rows x 128 bf16) via async global->LDS, zero-fill OOB rows
__device__ __forceinline__ void stage_A(bf16_t (*As)[GEMM_LDA],
                                        const bf16_t* __restrict__ A,
                                        int m0, int M, int t) {
    int row = t >> 1, half = t & 1;
    int gr  = m0 + row;
    unsigned lbase = (unsigned)(uintptr_t)&As[row][half * 64];
    if (gr < M) {
        const char* g = (const char*)(A + (size_t)gr * HID + half * 64);
#pragma unroll
        for (int i = 0; i < 8; ++i) async_copy16(lbase + i * 16, g + i * 16);
    } else {
        uint4 z = {0u, 0u, 0u, 0u};
        uint4* d4 = (uint4*)&As[row][half * 64];
#pragma unroll
        for (int i = 0; i < 8; ++i) d4[i] = z;
    }
}

// stage B^T (128 rows = output cols, 128 bf16 each) via async global->LDS
__device__ __forceinline__ void stage_B(bf16_t (*Bs)[GEMM_LDA],
                                        const bf16_t* __restrict__ Bt, int t) {
    unsigned lbase = (unsigned)(uintptr_t)&Bs[t][0];
    const char* g = (const char*)(Bt + (size_t)t * HID);
#pragma unroll
    for (int i = 0; i < 16; ++i) async_copy16(lbase + i * 16, g + i * 16);
}

// one full K=128 accumulation + store of a 64x128 tile
__device__ __forceinline__ void compute_store(bf16_t (*As)[GEMM_LDA],
                                              bf16_t (*Bs)[GEMM_LDA],
                                              float* __restrict__ C,
                                              int m0, int M, int wave, int lane) {
    const int half = lane >> 4;
    const int r16  = lane & 15;
    f32x8 acc[8] = {};
#pragma unroll
    for (int kk = 0; kk < HID; kk += 32) {
        Frag a;
        a.h[0] = *(const bf16x8*)&As[wave * 16 + r16][kk + half * 8];
        a.h[1] = *(const bf16x8*)&As[wave * 16 + r16][kk + 16 + half * 8];
#pragma unroll
        for (int n = 0; n < 8; ++n) {
            Frag b;
            b.h[0] = *(const bf16x8*)&Bs[n * 16 + r16][kk + half * 8];
            b.h[1] = *(const bf16x8*)&Bs[n * 16 + r16][kk + 16 + half * 8];
            acc[n] = __builtin_amdgcn_wmma_f32_16x16x32_bf16(
                false, a.v, false, b.v, (short)0, acc[n], false, false);
        }
    }
    const int lcol = lane & 15;
#pragma unroll
    for (int n = 0; n < 8; ++n) {
#pragma unroll
        for (int j = 0; j < 8; ++j) {
            int rowInTile = (lane < 16) ? j : (8 + j);
            int gr = m0 + wave * 16 + rowInTile;
            if (gr < M) C[(size_t)gr * HID + n * 16 + lcol] = acc[n][j];
        }
    }
}

// ---------------------------------------------------------------------------
// Single-B GEMM: C = A(bf16) * B (pre-transposed bf16)
// ---------------------------------------------------------------------------
__global__ __launch_bounds__(128)
void k_gemm_bf16_wmma(const bf16_t* __restrict__ A, const bf16_t* __restrict__ Bt,
                      float* __restrict__ C, int M) {
    __shared__ __align__(16) bf16_t As[GEMM_MT][GEMM_LDA];
    __shared__ __align__(16) bf16_t Bs[HID][GEMM_LDA];
    const int t = threadIdx.x, wave = t >> 5, lane = t & 31;
    const int m0 = blockIdx.x * GEMM_MT;

    stage_A(As, A, m0, M, t);
    stage_B(Bs, Bt, t);
    wait_async0();
    __syncthreads();
    compute_store(As, Bs, C, m0, M, wave, lane);
}

// ---------------------------------------------------------------------------
// Dual-B GEMM: A tile stays resident in LDS; two B phases (fs then fd).
// ---------------------------------------------------------------------------
__global__ __launch_bounds__(128)
void k_gemm_bf16_wmma_dual(const bf16_t* __restrict__ A,
                           const bf16_t* __restrict__ Bt1, const bf16_t* __restrict__ Bt2,
                           float* __restrict__ C1, float* __restrict__ C2, int M) {
    __shared__ __align__(16) bf16_t As[GEMM_MT][GEMM_LDA];
    __shared__ __align__(16) bf16_t Bs[HID][GEMM_LDA];
    const int t = threadIdx.x, wave = t >> 5, lane = t & 31;
    const int m0 = blockIdx.x * GEMM_MT;

    stage_A(As, A, m0, M, t);
    stage_B(Bs, Bt1, t);
    wait_async0();
    __syncthreads();
    compute_store(As, Bs, C1, m0, M, wave, lane);

    __syncthreads();            // all waves done reading Bs phase 1
    stage_B(Bs, Bt2, t);
    wait_async0();
    __syncthreads();
    compute_store(As, Bs, C2, m0, M, wave, lane);
}

// ---------------------------------------------------------------------------
// GATv2 edge scores: score[e,h] = sum_d attn[h,d]*leakyrelu(fs[src]+fd[dst])
// Also folds the segment-max (atomic) pass.
// ---------------------------------------------------------------------------
__global__ void k_edge_score_v2(const int* __restrict__ src, const int* __restrict__ dst,
                                const float* __restrict__ fs, const float* __restrict__ fd,
                                const float* __restrict__ attn,
                                float* __restrict__ score, float* __restrict__ mbuf, int E) {
    int tid = blockIdx.x * blockDim.x + threadIdx.x;
    if (tid >= E * NHEAD) return;
    int e = tid >> 2, h = tid & 3;
    int s = src[e], d = dst[e];
    const float4* ps = (const float4*)(fs + (size_t)s * HID + h * HDIM);
    const float4* pd = (const float4*)(fd + (size_t)d * HID + h * HDIM);
    const float4* pa = (const float4*)(attn + h * HDIM);
    float acc = 0.f;
#pragma unroll
    for (int i = 0; i < 8; ++i) {
        float4 x = ps[i], y = pd[i], w = pa[i];
        float v;
        v = x.x + y.x; v = v > 0.f ? v : 0.2f * v; acc += v * w.x;
        v = x.y + y.y; v = v > 0.f ? v : 0.2f * v; acc += v * w.y;
        v = x.z + y.z; v = v > 0.f ? v : 0.2f * v; acc += v * w.z;
        v = x.w + y.w; v = v > 0.f ? v : 0.2f * v; acc += v * w.w;
    }
    score[(size_t)e * NHEAD + h] = acc;
    atomicMaxF(&mbuf[(size_t)d * NHEAD + h], acc);
}

// ---------------------------------------------------------------------------
// GAT v1 edge scores: leakyrelu(el[src,h] + er[dst,h]) + segment max
// ---------------------------------------------------------------------------
__global__ void k_edge_score_v1(const int* __restrict__ src, const int* __restrict__ dst,
                                const float* __restrict__ el, const float* __restrict__ er,
                                float* __restrict__ score, float* __restrict__ mbuf, int E) {
    int tid = blockIdx.x * blockDim.x + threadIdx.x;
    if (tid >= E * NHEAD) return;
    int e = tid >> 2, h = tid & 3;
    int s = src[e], d = dst[e];
    float v = el[(size_t)s * NHEAD + h] + er[(size_t)d * NHEAD + h];
    v = v > 0.f ? v : 0.2f * v;
    score[(size_t)e * NHEAD + h] = v;
    atomicMaxF(&mbuf[(size_t)d * NHEAD + h], v);
}

// ---------------------------------------------------------------------------
// exp(score - m[dst]) in place + segment sum
// ---------------------------------------------------------------------------
__global__ void k_edge_exp(const int* __restrict__ dst, float* __restrict__ score,
                           const float* __restrict__ mbuf, float* __restrict__ sbuf, int E) {
    int tid = blockIdx.x * blockDim.x + threadIdx.x;
    if (tid >= E * NHEAD) return;
    int e = tid >> 2, h = tid & 3;
    int d = dst[e];
    float ex = __expf(score[(size_t)e * NHEAD + h] - mbuf[(size_t)d * NHEAD + h]);
    score[(size_t)e * NHEAD + h] = ex;
    atomicAdd(&sbuf[(size_t)d * NHEAD + h], ex);
}

// ---------------------------------------------------------------------------
// Aggregation: out[dst] += feat[src] * (ex / s[dst])  (one thread per edge and head, loops over 32 dims)
// ---------------------------------------------------------------------------
__global__ void k_edge_aggregate(const int* __restrict__ src, const int* __restrict__ dst,
                                 const float* __restrict__ score, const float* __restrict__ sbuf,
                                 const float* __restrict__ feat, float* __restrict__ out, int E) {
    int tid = blockIdx.x * blockDim.x + threadIdx.x;
    if (tid >= E * NHEAD) return;
    int e = tid >> 2, h = tid & 3;
    int s = src[e], d = dst[e];
    float alpha = score[(size_t)e * NHEAD + h] / sbuf[(size_t)d * NHEAD + h];
    const float* pf = feat + (size_t)s * HID + h * HDIM;
    float* po = out + (size_t)d * HID + h * HDIM;
#pragma unroll
    for (int i = 0; i < HDIM; ++i) atomicAdd(&po[i], pf[i] * alpha);
}

// ---------------------------------------------------------------------------
// Post-layer: bias + ELU, emit bf16 activations for the next GEMM
// ---------------------------------------------------------------------------
__global__ void k_bias_elu_bf16(const float* __restrict__ acc, const float* __restrict__ bias,
                                bf16_t* __restrict__ hnext, int N) {
    int tid = blockIdx.x * blockDim.x + threadIdx.x;
    int n = tid >> 7, c = tid & 127;
    if (n >= N) return;
    float v = acc[(size_t)n * HID + c] + bias[c];
    v = v > 0.f ? v : (__expf(v) - 1.f);
    hnext[(size_t)n * HID + c] = (bf16_t)v;
}

// ---------------------------------------------------------------------------
// Classifier node projections: el[n,h] = f[n,h,:].al[h,:], er likewise
// ---------------------------------------------------------------------------
__global__ void k_node_elr(const float* __restrict__ f, const float* __restrict__ al,
                           const float* __restrict__ ar, float* __restrict__ el,
                           float* __restrict__ er, int N) {
    int tid = blockIdx.x * blockDim.x + threadIdx.x;
    if (tid >= N * NHEAD) return;
    int n = tid >> 2, h = tid & 3;
    const float4* pf = (const float4*)(f + (size_t)n * HID + h * NCLS);
    const float4* pl = (const float4*)(al + h * NCLS);
    const float4* pr = (const float4*)(ar + h * NCLS);
    float sl = 0.f, sr = 0.f;
#pragma unroll
    for (int i = 0; i < 8; ++i) {
        float4 x = pf[i], a = pl[i], b = pr[i];
        sl += x.x * a.x + x.y * a.y + x.z * a.z + x.w * a.w;
        sr += x.x * b.x + x.y * b.y + x.z * b.z + x.w * b.w;
    }
    el[(size_t)n * NHEAD + h] = sl;
    er[(size_t)n * NHEAD + h] = sr;
}

// ---------------------------------------------------------------------------
// Final: add bias, mean over heads, softmax over 32 classes
// ---------------------------------------------------------------------------
__global__ void k_final_softmax(const float* __restrict__ acc, const float* __restrict__ bc,
                                float* __restrict__ y, int N) {
    int n = blockIdx.x * blockDim.x + threadIdx.x;
    if (n >= N) return;
    float g[NCLS];
#pragma unroll
    for (int v = 0; v < NCLS; ++v) {
        float t = 0.f;
#pragma unroll
        for (int h = 0; h < NHEAD; ++h)
            t += acc[(size_t)n * HID + h * NCLS + v] + bc[h * NCLS + v];
        g[v] = 0.25f * t;
    }
    float mx = g[0];
#pragma unroll
    for (int v = 1; v < NCLS; ++v) mx = fmaxf(mx, g[v]);
    float s = 0.f;
#pragma unroll
    for (int v = 0; v < NCLS; ++v) { g[v] = __expf(g[v] - mx); s += g[v]; }
    float inv = 1.f / s;
#pragma unroll
    for (int v = 0; v < NCLS; ++v) y[(size_t)n * NCLS + v] = g[v] * inv;
}

// ---------------------------------------------------------------------------
// Host-side orchestration
// ---------------------------------------------------------------------------
extern "C" void kernel_launch(void* const* d_in, const int* in_sizes, int n_in,
                              void* d_out, int out_size, void* d_ws, size_t ws_size,
                              hipStream_t stream) {
    const int*   in_feat = (const int*)d_in[0];
    const int*   src     = (const int*)d_in[1];
    const int*   dst     = (const int*)d_in[2];
    const float* emb     = (const float*)d_in[3];
    const float* W1s     = (const float*)d_in[4];
    const float* W1d     = (const float*)d_in[5];
    const float* a1      = (const float*)d_in[6];
    const float* b1      = (const float*)d_in[7];
    const float* W2s     = (const float*)d_in[8];
    const float* W2d     = (const float*)d_in[9];
    const float* a2      = (const float*)d_in[10];
    const float* b2      = (const float*)d_in[11];
    const float* Wc      = (const float*)d_in[12];
    const float* al      = (const float*)d_in[13];
    const float* ar      = (const float*)d_in[14];
    const float* bc      = (const float*)d_in[15];

    const int N = in_sizes[0];
    const int E = in_sizes[1];
    float* y = (float*)d_out;

    // --- workspace carve-up ---
    char* ws = (char*)d_ws;
    size_t off = 0;
    auto take = [&](size_t bytes) -> char* {
        char* p = ws + off;
        off = (off + bytes + 255) & ~(size_t)255;
        return p;
    };
    bf16_t* wbt    = (bf16_t*)take((size_t)5 * HID * HID * sizeof(bf16_t)); // 5 weights
    bf16_t* hbf    = (bf16_t*)take((size_t)N * HID * sizeof(bf16_t));
    float*  fs     = (float*) take((size_t)N * HID * sizeof(float));
    float*  fd     = (float*) take((size_t)N * HID * sizeof(float));
    float*  outacc = (float*) take((size_t)N * HID * sizeof(float));
    float*  score  = (float*) take((size_t)E * NHEAD * sizeof(float));
    float*  mbuf   = (float*) take((size_t)N * NHEAD * sizeof(float));
    float*  sbuf   = (float*) take((size_t)N * NHEAD * sizeof(float));
    float*  el     = (float*) take((size_t)N * NHEAD * sizeof(float));
    float*  er     = (float*) take((size_t)N * NHEAD * sizeof(float));

    bf16_t* bW1s = wbt + 0 * HID * HID;
    bf16_t* bW1d = wbt + 1 * HID * HID;
    bf16_t* bW2s = wbt + 2 * HID * HID;
    bf16_t* bW2d = wbt + 3 * HID * HID;
    bf16_t* bWc  = wbt + 4 * HID * HID;

    const int TB = 256;
    const int gW    = (HID * HID + TB - 1) / TB;
    const int gNC   = (N * HID + TB - 1) / TB;
    const int gEH   = (E * NHEAD + TB - 1) / TB;
    const int gNH   = (N * NHEAD + TB - 1) / TB;
    const int gN    = (N + TB - 1) / TB;
    const int gGemm = (N + GEMM_MT - 1) / GEMM_MT;

    // --- weight conversion (bf16, transposed) ---
    k_conv_weight<<<gW, TB, 0, stream>>>(W1s, bW1s);
    k_conv_weight<<<gW, TB, 0, stream>>>(W1d, bW1d);
    k_conv_weight<<<gW, TB, 0, stream>>>(W2s, bW2s);
    k_conv_weight<<<gW, TB, 0, stream>>>(W2d, bW2d);
    k_conv_weight<<<gW, TB, 0, stream>>>(Wc,  bWc);

    // --- embedding gather ---
    k_gather_emb<<<gNC, TB, 0, stream>>>(in_feat, emb, hbf, N);

    const float* attn[2] = {a1, a2};
    const float* bias[2] = {b1, b2};
    const bf16_t* Wsl[2] = {bW1s, bW2s};
    const bf16_t* Wdl[2] = {bW1d, bW2d};

    // --- two GATv2 layers ---
    for (int L = 0; L < 2; ++L) {
        k_gemm_bf16_wmma_dual<<<gGemm, 128, 0, stream>>>(hbf, Wsl[L], Wdl[L], fs, fd, N);
        k_fill_f32<<<gNH, TB, 0, stream>>>(mbuf, -1e30f, N * NHEAD);
        k_fill_f32<<<gNH, TB, 0, stream>>>(sbuf, 0.f, N * NHEAD);
        k_fill_f32<<<gNC, TB, 0, stream>>>(outacc, 0.f, N * HID);
        k_edge_score_v2<<<gEH, TB, 0, stream>>>(src, dst, fs, fd, attn[L], score, mbuf, E);
        k_edge_exp<<<gEH, TB, 0, stream>>>(dst, score, mbuf, sbuf, E);
        k_edge_aggregate<<<gEH, TB, 0, stream>>>(src, dst, score, sbuf, fs, outacc, E);
        k_bias_elu_bf16<<<gNC, TB, 0, stream>>>(outacc, bias[L], hbf, N);
    }

    // --- classifier GAT (v1) ---
    k_gemm_bf16_wmma<<<gGemm, 128, 0, stream>>>(hbf, bWc, fs, N);   // f = h @ Wc
    k_node_elr<<<gNH, TB, 0, stream>>>(fs, al, ar, el, er, N);
    k_fill_f32<<<gNH, TB, 0, stream>>>(mbuf, -1e30f, N * NHEAD);
    k_fill_f32<<<gNH, TB, 0, stream>>>(sbuf, 0.f, N * NHEAD);
    k_fill_f32<<<gNC, TB, 0, stream>>>(outacc, 0.f, N * HID);
    k_edge_score_v1<<<gEH, TB, 0, stream>>>(src, dst, el, er, score, mbuf, E);
    k_edge_exp<<<gEH, TB, 0, stream>>>(dst, score, mbuf, sbuf, E);
    k_edge_aggregate<<<gEH, TB, 0, stream>>>(src, dst, score, sbuf, fs, outacc, E);
    k_final_softmax<<<gN, TB, 0, stream>>>(outacc, bc, y, N);
}
